// BilinearFeedForward_27779848471445
// MI455X (gfx1250) — matrix operand order repaired and checked by reading, not compiled, not run on hardware
//
#include <hip/hip_runtime.h>

// ---------------------------------------------------------------------------
// BilinearFeedForward for MI455X (gfx1250), bf16 WMMA pipeline.
//   1) gemm_x2 (gate mode): Qb  = relu(x Wqr) * relu(x Wqi)        [bf16]
//   2) gemm_x2 (kv   mode): Kb  = x Wk ; Vb = x Wv                 [bf16]
//   3) colnorm:             invK[b,d] = 1/(||K[:,d]||+eps), same V [f32]
//   4) gemm_kv_attn:        KVb = relu( (Kb^T Vb) * invK_d*invV_e) [bf16]
//   5) gemm_out:            out = Qb @ KVb                          [f32]
// Double-buffered LDS staging, grouped fragment loads, async-LDS loads where
// the tile is a pure bf16 copy.
// ---------------------------------------------------------------------------

typedef __bf16 bf16;
typedef __bf16 v16bf __attribute__((ext_vector_type(16)));
typedef __bf16 v8bf  __attribute__((ext_vector_type(8)));
typedef __bf16 v2bf  __attribute__((ext_vector_type(2)));
typedef float  v8f   __attribute__((ext_vector_type(8)));

#define B_   8
#define N_   4096
#define D_   1024
#define BN_  (B_ * N_)
#define EPSILON 1e-5f

#define TM 128   // block tile rows
#define TN 64    // block tile cols
#define TK 32    // k-step (WMMA bf16 K)

#if defined(__AMDGCN__) && __has_builtin(__builtin_amdgcn_global_load_async_to_lds_b128)
#define ASYNC_LDS 1
typedef int v4i_ __attribute__((vector_size(16)));          // matches builtin proto
typedef __attribute__((address_space(3))) v4i_ lds_v4i;     // LDS-side pointee
#endif

__device__ __forceinline__ void wait_async0() {
#if defined(__AMDGCN__)
#if __has_builtin(__builtin_amdgcn_s_wait_asynccnt)
  __builtin_amdgcn_s_wait_asynccnt(0);
#else
  asm volatile("s_wait_asynccnt 0x0" ::: "memory");
#endif
#endif
}

__device__ __forceinline__ v8f wmma_bf16(v16bf a, v16bf b, v8f c) {
  return __builtin_amdgcn_wmma_f32_16x16x32_bf16(
      /*neg_a=*/false, a, /*neg_b=*/false, b,
      /*c_mod=*/(short)0, c, /*reuse_a=*/false, /*reuse_b=*/false);
}

__device__ __forceinline__ v16bf cat16(v8bf lo, v8bf hi) {
  v16bf r;
#pragma unroll
  for (int i = 0; i < 8; ++i) { r[i] = lo[i]; r[i + 8] = hi[i]; }
  return r;
}

// A fragment (16x32) from LDS tile [row][k], row stride TK.
__device__ __forceinline__ v16bf load_a(const bf16* sA, int rowBase, int lane) {
  const int m  = lane & 15;
  const int kb = (lane < 16) ? 0 : 8;
  const bf16* p = sA + (rowBase + m) * TK;
  v8bf lo = *(const v8bf*)(p + kb);
  v8bf hi = *(const v8bf*)(p + kb + 16);
  return cat16(lo, hi);
}

// B fragment (32x16) from LDS tile [col][k], col stride TK.
__device__ __forceinline__ v16bf load_b(const bf16* sB, int colBase, int lane) {
  const int n    = lane & 15;
  const int koff = (lane < 16) ? 0 : 8;
  const bf16* p = sB + (colBase + n) * TK + koff;
  v8bf lo = *(const v8bf*)(p);
  v8bf hi = *(const v8bf*)(p + 16);
  return cat16(lo, hi);
}

// ---------------------------------------------------------------------------
// Kernel 1/2: O = x @ {W0, W1} with fused epilogue.
// ---------------------------------------------------------------------------
__global__ __launch_bounds__(256) void gemm_x2(
    const float* __restrict__ X, const float* __restrict__ W0,
    const float* __restrict__ W1, bf16* __restrict__ O0,
    bf16* __restrict__ O1, int gateMode) {
  __shared__ __attribute__((aligned(16))) bf16 sA[2][TM * TK];
  __shared__ __attribute__((aligned(16))) bf16 sB[2][2][TN * TK];

  const int t = threadIdx.x;
  const int wave = t >> 5, lane = t & 31;
  const int rowBase = blockIdx.x * TM;
  const int colBase = blockIdx.y * TN;

  v8f acc0[4] = {}, acc1[4] = {};

  const int arow = t >> 1;
  const int akp  = (t & 1) * 16;
  const int wsel = t >> 7;
  const int u    = t & 127;
  const int wk0  = (u >> 3) * 2;   // even k row (pair wk0, wk0+1)
  const int wnp  = (u & 7) * 8;    // 8 cols per thread
  const float* Wm = wsel ? W1 : W0;

  float4 fa[4];  // staged A: 16 f32
  float4 fb[4];  // staged B: 2 k-rows x 8 f32

  auto loadA = [&](int kt) {
    const float* src = X + (size_t)(rowBase + arow) * D_ + kt + akp;
    fa[0] = ((const float4*)src)[0];
    fa[1] = ((const float4*)src)[1];
    fa[2] = ((const float4*)src)[2];
    fa[3] = ((const float4*)src)[3];
    if (kt + 2 * TK < D_) __builtin_prefetch(src + 2 * TK, 0, 3);
  };
  auto loadB = [&](int kt) {
    const float* s0 = Wm + (size_t)(kt + wk0) * D_ + colBase + wnp;
    const float* s1 = s0 + D_;
    fb[0] = ((const float4*)s0)[0];
    fb[1] = ((const float4*)s0)[1];
    fb[2] = ((const float4*)s1)[0];
    fb[3] = ((const float4*)s1)[1];
  };
  auto storeA = [&](int p) {
    v8bf p0, p1;
    p0[0]=(bf16)fa[0].x; p0[1]=(bf16)fa[0].y; p0[2]=(bf16)fa[0].z; p0[3]=(bf16)fa[0].w;
    p0[4]=(bf16)fa[1].x; p0[5]=(bf16)fa[1].y; p0[6]=(bf16)fa[1].z; p0[7]=(bf16)fa[1].w;
    p1[0]=(bf16)fa[2].x; p1[1]=(bf16)fa[2].y; p1[2]=(bf16)fa[2].z; p1[3]=(bf16)fa[2].w;
    p1[4]=(bf16)fa[3].x; p1[5]=(bf16)fa[3].y; p1[6]=(bf16)fa[3].z; p1[7]=(bf16)fa[3].w;
    *(v8bf*)(&sA[p][arow * TK + akp])     = p0;
    *(v8bf*)(&sA[p][arow * TK + akp + 8]) = p1;
  };
  auto storeB = [&](int p) {
    const float r0[8] = {fb[0].x, fb[0].y, fb[0].z, fb[0].w,
                         fb[1].x, fb[1].y, fb[1].z, fb[1].w};
    const float r1[8] = {fb[2].x, fb[2].y, fb[2].z, fb[2].w,
                         fb[3].x, fb[3].y, fb[3].z, fb[3].w};
#pragma unroll
    for (int j = 0; j < 8; ++j) {
      v2bf pk;
      pk[0] = (bf16)r0[j];   // k = wk0
      pk[1] = (bf16)r1[j];   // k = wk0+1
      *(v2bf*)(&sB[p][wsel][(wnp + j) * TK + wk0]) = pk;
    }
  };

  loadA(0); loadB(0); storeA(0); storeB(0);
  int p = 0;
  for (int kt = 0; kt < D_; kt += TK, p ^= 1) {
    const bool more = (kt + TK) < D_;
    if (more) { loadA(kt + TK); loadB(kt + TK); }
    __syncthreads();
    v16bf a = load_a(sA[p], wave * 16, lane);
    v16bf b0[4], b1[4];
#pragma unroll
    for (int j = 0; j < 4; ++j) {
      b0[j] = load_b(sB[p][0], j * 16, lane);
      b1[j] = load_b(sB[p][1], j * 16, lane);
    }
#pragma unroll
    for (int j = 0; j < 4; ++j) {
      acc0[j] = wmma_bf16(a, b0[j], acc0[j]);
      acc1[j] = wmma_bf16(a, b1[j], acc1[j]);
    }
    if (more) { storeA(p ^ 1); storeB(p ^ 1); }
  }

  const int m0 = ((lane >> 4) & 1) * 8;
  const int nn = lane & 15;
#pragma unroll
  for (int j = 0; j < 4; ++j) {
    const int col = colBase + j * 16 + nn;
#pragma unroll
    for (int r = 0; r < 8; ++r) {
      const int row = rowBase + wave * 16 + m0 + r;
      const size_t idx = (size_t)row * D_ + col;
      const float v0 = acc0[j][r], v1 = acc1[j][r];
      if (gateMode) {
        O0[idx] = (bf16)(fmaxf(v0, 0.f) * fmaxf(v1, 0.f));
      } else {
        O0[idx] = (bf16)v0;
        O1[idx] = (bf16)v1;
      }
    }
  }
}

// ---------------------------------------------------------------------------
// Kernel 3: deterministic per-(b,d) inverse L2 norms over sequence axis.
// ---------------------------------------------------------------------------
__global__ __launch_bounds__(256) void colnorm(
    const bf16* __restrict__ K, const bf16* __restrict__ V,
    float* __restrict__ invK, float* __restrict__ invV) {
  const int b  = blockIdx.x >> 4;
  const int d0 = (blockIdx.x & 15) * 64;
  const int t  = threadIdx.x;
  const int d  = d0 + (t & 63);
  const int n0 = t >> 6;
  const size_t base = (size_t)b * N_ * D_ + d;
  float sk = 0.f, sv = 0.f;
  for (int n = n0; n < N_; n += 4) {
    const float kv = (float)K[base + (size_t)n * D_];
    const float vv = (float)V[base + (size_t)n * D_];
    sk += kv * kv;
    sv += vv * vv;
  }
  __shared__ float rk[256], rv[256];
  rk[t] = sk; rv[t] = sv;
  __syncthreads();
  if (t < 64) {
    sk = rk[t] + rk[t + 64] + rk[t + 128] + rk[t + 192];
    sv = rv[t] + rv[t + 64] + rv[t + 128] + rv[t + 192];
    invK[b * D_ + d0 + t] = 1.0f / (sqrtf(sk) + EPSILON);
    invV[b * D_ + d0 + t] = 1.0f / (sqrtf(sv) + EPSILON);
  }
}

// ---------------------------------------------------------------------------
// Kernel 4: KV[b,d,e] = relu( sum_n K[b,n,d]*V[b,n,e] * invK[b,d]*invV[b,e] )
// ---------------------------------------------------------------------------
__global__ __launch_bounds__(256) void gemm_kv_attn(
    const bf16* __restrict__ K, const bf16* __restrict__ V,
    const float* __restrict__ invK, const float* __restrict__ invV,
    bf16* __restrict__ KV) {
  __shared__ __attribute__((aligned(16))) bf16 sA[2][TM * TK];  // [d][n]
  __shared__ __attribute__((aligned(16))) bf16 sB[2][TN * TK];  // [e][n]
  const int t = threadIdx.x, wave = t >> 5, lane = t & 31;
  const int dBase = blockIdx.x * TM;
  const int eBase = blockIdx.y * TN;
  const int b = blockIdx.z;
  const size_t boff = (size_t)b * N_ * D_;

  v8f acc[4] = {};

  const int srcn = t >> 3;        // n row inside tile
  const int adp  = (t & 7) * 16;  // A: 16 d's per thread
  const int bep  = (t & 7) * 8;   // B: 8 e's per thread

  v8bf ka0, ka1, vb0;
  auto loadT = [&](int nt) {
    const bf16* ks = K + boff + (size_t)(nt + srcn) * D_ + dBase + adp;
    ka0 = ((const v8bf*)ks)[0];
    ka1 = ((const v8bf*)ks)[1];
    const bf16* vs = V + boff + (size_t)(nt + srcn) * D_ + eBase + bep;
    vb0 = *(const v8bf*)vs;
    if (nt + 2 * TK < N_) __builtin_prefetch(ks + (size_t)TK * D_, 0, 3);
  };
  auto storeT = [&](int p) {
#pragma unroll
    for (int j = 0; j < 8; ++j) {
      sA[p][(adp + j) * TK + srcn]     = ka0[j];
      sA[p][(adp + 8 + j) * TK + srcn] = ka1[j];
    }
#pragma unroll
    for (int j = 0; j < 8; ++j) sB[p][(bep + j) * TK + srcn] = vb0[j];
  };

  loadT(0); storeT(0);
  int p = 0;
  for (int nt = 0; nt < N_; nt += TK, p ^= 1) {
    const bool more = (nt + TK) < N_;
    if (more) loadT(nt + TK);
    __syncthreads();
    v16bf a = load_a(sA[p], wave * 16, lane);
    v16bf bb[4];
#pragma unroll
    for (int j = 0; j < 4; ++j) bb[j] = load_b(sB[p], j * 16, lane);
#pragma unroll
    for (int j = 0; j < 4; ++j) acc[j] = wmma_bf16(a, bb[j], acc[j]);
    if (more) storeT(p ^ 1);
  }

  const int m0 = ((lane >> 4) & 1) * 8;
  const int nn = lane & 15;
#pragma unroll
  for (int j = 0; j < 4; ++j) {
    const int e = eBase + j * 16 + nn;
    const float se = invV[b * D_ + e];
#pragma unroll
    for (int r = 0; r < 8; ++r) {
      const int d = dBase + wave * 16 + m0 + r;
      const float val = acc[j][r] * (invK[b * D_ + d] * se);
      KV[((size_t)b * D_ + d) * D_ + e] = (bf16)fmaxf(val, 0.f);
    }
  }
}

// ---------------------------------------------------------------------------
// Kernel 5: out[b,n,e] = sum_d Q[b,n,d] * KV[b,d,e]. A-tile streamed with
// GLOBAL_LOAD_ASYNC_TO_LDS (ASYNCcnt) when the toolchain exposes it.
// ---------------------------------------------------------------------------
__global__ __launch_bounds__(256) void gemm_out(
    const bf16* __restrict__ Q, const bf16* __restrict__ KV,
    float* __restrict__ O) {
  __shared__ __attribute__((aligned(16))) bf16 sA[2][TM * TK];  // [row][k]
  __shared__ __attribute__((aligned(16))) bf16 sB[2][TN * TK];  // [col][k]
  const int t = threadIdx.x, wave = t >> 5, lane = t & 31;
  const int rowBase = blockIdx.x * TM;
  const int colBase = blockIdx.y * TN;
  const int b = rowBase >> 12;  // N_ = 4096 rows per batch
  const bf16* KVb = KV + (size_t)b * D_ * D_;

  v8f acc[4] = {};

  const int arow  = t >> 1, akp = (t & 1) * 16;
  const int bkrow = t >> 3, bep = (t & 7) * 8;

  v8bf qb0;
  auto loadB = [&](int kt) {
    const bf16* src = KVb + (size_t)(kt + bkrow) * D_ + colBase + bep;
    qb0 = *(const v8bf*)src;
  };
  auto storeB = [&](int p) {
#pragma unroll
    for (int j = 0; j < 8; ++j) sB[p][(bep + j) * TK + bkrow] = qb0[j];
  };

#if defined(ASYNC_LDS)
  auto asyncA = [&](int kt, int p) {
    const bf16* src = Q + (size_t)(rowBase + arow) * D_ + kt + akp;
    bf16* dst = &sA[p][arow * TK + akp];
    __builtin_amdgcn_global_load_async_to_lds_b128((v4i_*)src, (lds_v4i*)dst,
                                                   0, 0);
    __builtin_amdgcn_global_load_async_to_lds_b128((v4i_*)(src + 8),
                                                   (lds_v4i*)(dst + 8), 0, 0);
  };
  asyncA(0, 0);
#else
  v8bf qa0, qa1;
  auto loadA = [&](int kt) {
    const bf16* src = Q + (size_t)(rowBase + arow) * D_ + kt + akp;
    qa0 = ((const v8bf*)src)[0];
    qa1 = ((const v8bf*)src)[1];
  };
  auto storeA = [&](int p) {
    *(v8bf*)(&sA[p][arow * TK + akp])     = qa0;
    *(v8bf*)(&sA[p][arow * TK + akp + 8]) = qa1;
  };
  loadA(0); storeA(0);
#endif
  loadB(0); storeB(0);

  int p = 0;
  for (int kt = 0; kt < D_; kt += TK, p ^= 1) {
    const bool more = (kt + TK) < D_;
    if (more) loadB(kt + TK);
#if defined(ASYNC_LDS)
    wait_async0();                       // our async tile stores complete
    __syncthreads();                     // everyone done reading sA[p^1]
    if (more) asyncA(kt + TK, p ^ 1);    // fill next buffer during compute
#else
    if (more) loadA(kt + TK);
    __syncthreads();
#endif
    v16bf a = load_a(sA[p], wave * 16, lane);
    v16bf bb[4];
#pragma unroll
    for (int j = 0; j < 4; ++j) bb[j] = load_b(sB[p], j * 16, lane);
#pragma unroll
    for (int j = 0; j < 4; ++j) acc[j] = wmma_bf16(a, bb[j], acc[j]);
    if (more) {
      storeB(p ^ 1);
#if !defined(ASYNC_LDS)
      storeA(p ^ 1);
#endif
    }
  }

  const int m0 = ((lane >> 4) & 1) * 8;
  const int nn = lane & 15;
#pragma unroll
  for (int j = 0; j < 4; ++j) {
    const int col = colBase + j * 16 + nn;
#pragma unroll
    for (int r = 0; r < 8; ++r) {
      const int row = rowBase + wave * 16 + m0 + r;
      O[(size_t)row * D_ + col] = acc[j][r];
    }
  }
}

// ---------------------------------------------------------------------------
extern "C" void kernel_launch(void* const* d_in, const int* in_sizes, int n_in,
                              void* d_out, int out_size, void* d_ws,
                              size_t ws_size, hipStream_t stream) {
  (void)in_sizes; (void)n_in; (void)out_size; (void)ws_size;
  const float* x   = (const float*)d_in[0];
  const float* wqr = (const float*)d_in[1];
  const float* wqi = (const float*)d_in[2];
  const float* wk  = (const float*)d_in[3];
  const float* wv  = (const float*)d_in[4];
  float* out = (float*)d_out;

  char* ws = (char*)d_ws;
  const size_t xElems = (size_t)BN_ * D_;
  bf16*  Qb   = (bf16*)(ws);
  bf16*  Kb   = (bf16*)(ws + xElems * 2);
  bf16*  Vb   = (bf16*)(ws + xElems * 4);
  bf16*  KVb  = (bf16*)(ws + xElems * 6);
  float* invK = (float*)(ws + xElems * 6 + (size_t)B_ * D_ * D_ * 2);
  float* invV = invK + B_ * D_;

  dim3 blk(256);
  dim3 gRow(BN_ / TM, D_ / TN);     // 256 x 16
  dim3 gKV(D_ / TM, D_ / TN, B_);   // 8 x 16 x 8

  gemm_x2<<<gRow, blk, 0, stream>>>(x, wqr, wqi, Qb, Qb, 1);
  gemm_x2<<<gRow, blk, 0, stream>>>(x, wk, wv, Kb, Vb, 0);
  colnorm<<<dim3(B_ * D_ / 64), blk, 0, stream>>>(Kb, Vb, invK, invV);
  gemm_kv_attn<<<gKV, blk, 0, stream>>>(Kb, Vb, invK, invV, KVb);
  gemm_out<<<gRow, blk, 0, stream>>>(Qb, KVb, out);
}
